// WassersteinBarycenterTransport_89446988906814
// MI455X (gfx1250) — matrix operand order repaired and checked by reading, not compiled, not run on hardware
//
#include <hip/hip_runtime.h>
#include <hip/hip_bf16.h>

// ---------------------------------------------------------------------------
// Wasserstein barycenter transport for MI455X (gfx1250, wave32, WMMA).
// GEMMs: v_wmma_f32_16x16x32_bf16 with b128 fragment loads.
// Sinkhorn: persistent cooperative kernel, LDS-staged pre-scaled potentials,
// v_fma + v_exp(base2) per element over the L2-resident 64MB cost matrix.
// ---------------------------------------------------------------------------

#define NN 4096
#define DD 64
#define NCP 32          // label columns padded 10 -> 32 (one full bf16 K-chunk)
#define NCL 10
#define KDOM 3
#define REG 0.1f
#define INVREG 10.0f
#define NITER_SINK 100
#define NITER_BAR 10
#define SGRID 128       // persistent sinkhorn grid (blocks of 256)

#define LOG2E 1.4426950408889634f
#define LN2   0.6931471805599453f

typedef __attribute__((ext_vector_type(16))) __bf16          v16bf;
typedef __attribute__((ext_vector_type(8)))  float           v8f;
typedef __attribute__((ext_vector_type(4)))  float           f4;
typedef __attribute__((ext_vector_type(8)))  unsigned short  us8;

union BF16x16 { v16bf v; unsigned short u[16]; unsigned w[8]; us8 h[2]; };

__device__ inline unsigned short f2bf(float x) {
    unsigned u = __float_as_uint(x);
    u += 0x7FFFu + ((u >> 16) & 1u);          // round-to-nearest-even
    return (unsigned short)(u >> 16);
}

// pack two positive floats to bf16 pair (round-half-up) with one v_perm_b32
__device__ inline unsigned pack2bf(float lo, float hi) {
    return __builtin_amdgcn_perm(__float_as_uint(hi) + 0x8000u,
                                 __float_as_uint(lo) + 0x8000u, 0x07060302u);
}

#define WMMA_BF16(A, B, C) \
    __builtin_amdgcn_wmma_f32_16x16x32_bf16(false, (A), false, (B), (short)0, (C), false, false)

// --------------------------- utility kernels -------------------------------

__global__ void zero_kernel(float* p, int n) {
    int i = blockIdx.x * blockDim.x + threadIdx.x;
    if (i < n) p[i] = 0.f;
}

__global__ void scale_kernel(const float* src, float* dst, float sc, int n) {
    int i = blockIdx.x * blockDim.x + threadIdx.x;
    if (i < n) dst[i] = src[i] * sc;
}

__global__ void reset_scal_kernel(int* scal) {
    if (blockIdx.x == 0 && threadIdx.x < 8) scal[threadIdx.x] = 0;
}

// f32 [NN x src_cols] -> bf16 [NN x dst_cols] (zero-padded) + row sq-norms.
__global__ void cvt_pad_norm_kernel(const float* __restrict__ src, int src_cols,
                                    unsigned short* __restrict__ dst, int dst_cols,
                                    float* __restrict__ norms) {
    int wid  = (blockIdx.x * blockDim.x + threadIdx.x) >> 5;
    int lane = threadIdx.x & 31;
    float s = 0.f;
    for (int c = lane; c < dst_cols; c += 32) {
        float x = (c < src_cols) ? src[(size_t)wid * src_cols + c] : 0.f;
        dst[(size_t)wid * dst_cols + c] = f2bf(x);
        s += x * x;
    }
    for (int off = 16; off; off >>= 1) s += __shfl_xor(s, off, 32);
    if (lane == 0) norms[wid] = s;
}

// f32 [NN x src_cols] -> bf16 transposed [dst_rows x NN] (rows >= src_cols zero)
__global__ void transpose_cvt_kernel(const float* __restrict__ src, int src_cols,
                                     unsigned short* __restrict__ dst, int dst_rows) {
    int tid = blockIdx.x * blockDim.x + threadIdx.x;
    if (tid >= NN * dst_rows) return;
    int r  = tid & (NN - 1);
    int dr = tid >> 12;                 // / NN
    float x = (dr < src_cols) ? src[(size_t)r * src_cols + dr] : 0.f;
    dst[(size_t)dr * NN + r] = f2bf(x);
}

__global__ void pad_labels_f32_kernel(const float* __restrict__ src, float* __restrict__ dst, int rows) {
    int r = blockIdx.x * blockDim.x + threadIdx.x;
    if (r >= rows) return;
    for (int c = 0; c < 16; ++c)
        dst[(size_t)r * 16 + c] = (c < NCL) ? src[(size_t)r * NCL + c] : 0.f;
}

__global__ void copy_labels_out_kernel(const float* __restrict__ src16, float* __restrict__ out10, int rows) {
    int r = blockIdx.x * blockDim.x + threadIdx.x;
    if (r >= rows) return;
    for (int c = 0; c < NCL; ++c)
        out10[(size_t)r * NCL + c] = src16[(size_t)r * 16 + c];
}

// --------------------- WMMA squared-distance kernel ------------------------
// C_out[i,j] = (ADDIN? Cin + lw*d : d), d = max(0, an[i]+bn[j]-2*A_i.B_j)
template <int KDIM, int TJ, bool ADDIN>
__global__ void __launch_bounds__(256) dist_wmma_kernel(
    const unsigned short* __restrict__ Abf, const float* __restrict__ an,
    const unsigned short* __restrict__ Bbf, const float* __restrict__ bn,
    float* __restrict__ Cout, int* __restrict__ maxbits,
    const float* __restrict__ Cin, const int* __restrict__ lwbits) {
    constexpr int CH = KDIM / 32;
    constexpr int GJ = NN / 16 / TJ;
    int wid  = (blockIdx.x * blockDim.x + threadIdx.x) >> 5;
    int lane = threadIdx.x & 31;
    int ti  = wid / GJ;
    int tj0 = (wid % GJ) * TJ;
    int mn = lane & 15, half = lane >> 4;

    BF16x16 a[CH];
    const unsigned short* Arow = Abf + (size_t)(ti * 16 + mn) * KDIM;
#pragma unroll
    for (int c = 0; c < CH; ++c) {
        a[c].h[0] = *(const us8*)(Arow + c * 32 + half * 8);
        a[c].h[1] = *(const us8*)(Arow + c * 32 + 16 + half * 8);
    }
    v8f acc[TJ];
    v8f zf = {0.f, 0.f, 0.f, 0.f, 0.f, 0.f, 0.f, 0.f};
#pragma unroll
    for (int t = 0; t < TJ; ++t) acc[t] = zf;
#pragma unroll
    for (int t = 0; t < TJ; ++t) {
        const unsigned short* Brow = Bbf + (size_t)((tj0 + t) * 16 + mn) * KDIM;
#pragma unroll
        for (int c = 0; c < CH; ++c) {
            BF16x16 b;
            b.h[0] = *(const us8*)(Brow + c * 32 + half * 8);
            b.h[1] = *(const us8*)(Brow + c * 32 + 16 + half * 8);
            acc[t] = WMMA_BF16(a[c].v, b.v, acc[t]);
        }
    }
    float lw = ADDIN ? __int_as_float(*lwbits) : 0.f;
    float lmax = 0.f;
#pragma unroll
    for (int t = 0; t < TJ; ++t) {
        int j = (tj0 + t) * 16 + mn;
        float bnj = bn[j];
#pragma unroll
        for (int r = 0; r < 8; ++r) {
            int i = ti * 16 + half * 8 + r;
            float d = fmaxf(an[i] + bnj - 2.f * acc[t][r], 0.f);
            float val = ADDIN ? (Cin[(size_t)i * NN + j] + lw * d) : d;
            Cout[(size_t)i * NN + j] = val;
            lmax = fmaxf(lmax, val);
        }
    }
    for (int off = 16; off; off >>= 1) lmax = fmaxf(lmax, __shfl_xor(lmax, off, 32));
    if (lane == 0) atomicMax(maxbits, __float_as_int(lmax));   // vals >= 0
}

// ------------------- persistent cooperative Sinkhorn -----------------------
__device__ inline void grid_sync(volatile int* bar) {
    __syncthreads();
    if (threadIdx.x == 0) {
        int gen = bar[1];
        __threadfence();
        if (atomicAdd((int*)&bar[0], 1) == SGRID - 1) {
            bar[0] = 0;
            __threadfence();
            atomicAdd((int*)&bar[1], 1);
        } else {
            while (bar[1] == gen) { __builtin_amdgcn_s_sleep(2); }
            __threadfence();       // acquire other blocks' f/g writes
        }
    }
    __syncthreads();
}

// C normalized to [0,1]; potentials O(1) => exponents ~[-30,30]: plain
// sum(exp2) + log2 is safe. Potentials staged in LDS pre-scaled by
// INVREG*log2e so the inner loop is 1 v_fma + 1 v_exp + 1 v_add per element.
__global__ void __launch_bounds__(256, 1) sinkhorn_kernel(
    const float* __restrict__ C, float* __restrict__ f, float* __restrict__ g,
    const int* __restrict__ maxbits, int* bar, int niter) {
    const float log_ab = -8.31776616672f;   // ln(1/4096)
    const float IL = INVREG * LOG2E;
    __shared__ __align__(16) float sbuf[NN];    // staged scaled g / f
    __shared__ float spart[8][32];
    int lane = threadIdx.x & 31, wv = threadIdx.x >> 5;
    int gtid = blockIdx.x * blockDim.x + threadIdx.x;
    if (gtid < NN) { f[gtid] = 0.f; g[gtid] = 0.f; }
    grid_sync(bar);
    float csr = (1.0f / __int_as_float(*maxbits)) * IL;   // C -> exponent scale

    for (int it = 0; it < niter; ++it) {
        // stage g*IL in LDS (vectorized)
        for (int v = threadIdx.x; v < NN / 4; v += 256) {
            f4 t = *(const f4*)(g + v * 4);
            *(f4*)(&sbuf[v * 4]) = t * IL;
        }
        __syncthreads();
        // f-update: one wave per row
        for (int p = 0; p < 4; ++p) {
            int row = blockIdx.x * 32 + p * 8 + wv;
            const float* Crow = C + (size_t)row * NN;
            float s = 0.f;
            for (int t = 0; t < NN / 128; ++t) {
                int j = t * 128 + lane * 4;
                f4 cv = *(const f4*)(Crow + j);
                f4 gv = *(const f4*)(&sbuf[j]);
#pragma unroll
                for (int q = 0; q < 4; ++q)
                    s += __builtin_amdgcn_exp2f(fmaf(cv[q], -csr, gv[q]));
            }
            for (int off = 16; off; off >>= 1) s += __shfl_xor(s, off, 32);
            if (lane == 0) f[row] = REG * (log_ab - LN2 * __builtin_amdgcn_logf(s));
        }
        grid_sync(bar);
        // stage f*IL in LDS
        for (int v = threadIdx.x; v < NN / 4; v += 256) {
            f4 t = *(const f4*)(f + v * 4);
            *(f4*)(&sbuf[v * 4]) = t * IL;
        }
        __syncthreads();
        // g-update: block owns 32 columns, lane-per-column (coalesced b32)
        {
            int col = blockIdx.x * 32 + lane;
            float s = 0.f;
            int r0 = wv * (NN / 8), r1 = r0 + NN / 8;
            for (int r = r0; r < r1; ++r)
                s += __builtin_amdgcn_exp2f(fmaf(C[(size_t)r * NN + col], -csr, sbuf[r]));
            spart[wv][lane] = s;
            __syncthreads();
            if (wv == 0) {
                float S = 0.f;
#pragma unroll
                for (int w = 0; w < 8; ++w) S += spart[w][lane];
                g[col] = REG * (log_ab - LN2 * __builtin_amdgcn_logf(S));
            }
            __syncthreads();
        }
        grid_sync(bar);
    }
}

// ------- fused plan construction + barycentric projection (WMMA) -----------
__global__ void __launch_bounds__(256) proj_wmma_kernel(
    const float* __restrict__ C, const float* __restrict__ f, const float* __restrict__ g,
    const int* __restrict__ maxbits,
    const unsigned short* __restrict__ XT,   // [DD x NN]
    const unsigned short* __restrict__ YT,   // [16 x NN]
    float* __restrict__ Xacc, float* __restrict__ Yacc) {
    const float IL = INVREG * LOG2E;
    int ti   = (blockIdx.x * blockDim.x + threadIdx.x) >> 5;   // 256 row tiles
    int lane = threadIdx.x & 31;
    int mn = lane & 15, half = lane >> 4;
    float csr = (1.f / __int_as_float(*maxbits)) * IL;
    float fm2 = f[ti * 16 + mn] * IL;
    const float* Crow = C + (size_t)(ti * 16 + mn) * NN;
    v8f zf = {0.f, 0.f, 0.f, 0.f, 0.f, 0.f, 0.f, 0.f};
    v8f ax[4]; v8f ay = zf;
#pragma unroll
    for (int t = 0; t < 4; ++t) ax[t] = zf;

    for (int c = 0; c < NN / 32; ++c) {
        int j0 = c * 32 + half * 8;
        f4 c0 = *(const f4*)(Crow + j0);
        f4 c1 = *(const f4*)(Crow + j0 + 4);
        f4 c2 = *(const f4*)(Crow + j0 + 16);
        f4 c3 = *(const f4*)(Crow + j0 + 20);
        f4 g0 = *(const f4*)(g + j0);
        f4 g1 = *(const f4*)(g + j0 + 4);
        f4 g2 = *(const f4*)(g + j0 + 16);
        f4 g3 = *(const f4*)(g + j0 + 20);
        float e[16];
#pragma unroll
        for (int q = 0; q < 4; ++q) {
            e[q]      = __builtin_amdgcn_exp2f(fmaf(c0[q], -csr, fmaf(g0[q], IL, fm2)));
            e[4 + q]  = __builtin_amdgcn_exp2f(fmaf(c1[q], -csr, fmaf(g1[q], IL, fm2)));
            e[8 + q]  = __builtin_amdgcn_exp2f(fmaf(c2[q], -csr, fmaf(g2[q], IL, fm2)));
            e[12 + q] = __builtin_amdgcn_exp2f(fmaf(c3[q], -csr, fmaf(g3[q], IL, fm2)));
        }
        BF16x16 a;
#pragma unroll
        for (int q = 0; q < 8; ++q) a.w[q] = pack2bf(e[2 * q], e[2 * q + 1]);

        int kb = c * 32 + half * 8;
#pragma unroll
        for (int ct = 0; ct < 4; ++ct) {
            BF16x16 b;
            const unsigned short* bp = XT + (size_t)(ct * 16 + mn) * NN + kb;
            b.h[0] = *(const us8*)bp; b.h[1] = *(const us8*)(bp + 16);
            ax[ct] = WMMA_BF16(a.v, b.v, ax[ct]);
        }
        BF16x16 b;
        const unsigned short* yp = YT + (size_t)mn * NN + kb;
        b.h[0] = *(const us8*)yp; b.h[1] = *(const us8*)(yp + 16);
        ay = WMMA_BF16(a.v, b.v, ay);
    }
#pragma unroll
    for (int r = 0; r < 8; ++r) {
        int i = ti * 16 + half * 8 + r;
#pragma unroll
        for (int ct = 0; ct < 4; ++ct)
            Xacc[(size_t)i * DD + ct * 16 + mn] += ax[ct][r];
        Yacc[(size_t)i * 16 + mn] += ay[r];
    }
}

// row sums of the final plan (one wave per row)
__global__ void rowsum_kernel(const float* __restrict__ C, const float* __restrict__ f,
                              const float* __restrict__ g, const int* __restrict__ maxbits,
                              float* __restrict__ rs) {
    const float IL = INVREG * LOG2E;
    int row  = (blockIdx.x * blockDim.x + threadIdx.x) >> 5;
    int lane = threadIdx.x & 31;
    float csr = (1.f / __int_as_float(*maxbits)) * IL;
    const float* Crow = C + (size_t)row * NN;
    float fm2 = f[row] * IL, s = 0.f;
    for (int t = 0; t < NN / 128; ++t) {
        int j = t * 128 + lane * 4;
        f4 cv = *(const f4*)(Crow + j);
        f4 gv = *(const f4*)(g + j);
#pragma unroll
        for (int q = 0; q < 4; ++q)
            s += __builtin_amdgcn_exp2f(fmaf(cv[q], -csr, fmaf(gv[q], IL, fm2)));
    }
    for (int off = 16; off; off >>= 1) s += __shfl_xor(s, off, 32);
    if (lane == 0) rs[row] = s;
}

// final transport: out[i,:] = (sum_j pi_ij * Xt[j,:]) / rs[i]
__global__ void __launch_bounds__(256) trans_wmma_kernel(
    const float* __restrict__ C, const float* __restrict__ f, const float* __restrict__ g,
    const int* __restrict__ maxbits, const unsigned short* __restrict__ XtT,  // [DD x NN]
    const float* __restrict__ rs, float* __restrict__ out) {
    const float IL = INVREG * LOG2E;
    int ti   = (blockIdx.x * blockDim.x + threadIdx.x) >> 5;
    int lane = threadIdx.x & 31;
    int mn = lane & 15, half = lane >> 4;
    float csr = (1.f / __int_as_float(*maxbits)) * IL;
    float fm2 = f[ti * 16 + mn] * IL;
    const float* Crow = C + (size_t)(ti * 16 + mn) * NN;
    v8f zf = {0.f, 0.f, 0.f, 0.f, 0.f, 0.f, 0.f, 0.f};
    v8f ax[4];
#pragma unroll
    for (int t = 0; t < 4; ++t) ax[t] = zf;

    for (int c = 0; c < NN / 32; ++c) {
        int j0 = c * 32 + half * 8;
        f4 c0 = *(const f4*)(Crow + j0);
        f4 c1 = *(const f4*)(Crow + j0 + 4);
        f4 c2 = *(const f4*)(Crow + j0 + 16);
        f4 c3 = *(const f4*)(Crow + j0 + 20);
        f4 g0 = *(const f4*)(g + j0);
        f4 g1 = *(const f4*)(g + j0 + 4);
        f4 g2 = *(const f4*)(g + j0 + 16);
        f4 g3 = *(const f4*)(g + j0 + 20);
        float e[16];
#pragma unroll
        for (int q = 0; q < 4; ++q) {
            e[q]      = __builtin_amdgcn_exp2f(fmaf(c0[q], -csr, fmaf(g0[q], IL, fm2)));
            e[4 + q]  = __builtin_amdgcn_exp2f(fmaf(c1[q], -csr, fmaf(g1[q], IL, fm2)));
            e[8 + q]  = __builtin_amdgcn_exp2f(fmaf(c2[q], -csr, fmaf(g2[q], IL, fm2)));
            e[12 + q] = __builtin_amdgcn_exp2f(fmaf(c3[q], -csr, fmaf(g3[q], IL, fm2)));
        }
        BF16x16 a;
#pragma unroll
        for (int q = 0; q < 8; ++q) a.w[q] = pack2bf(e[2 * q], e[2 * q + 1]);

        int kb = c * 32 + half * 8;
#pragma unroll
        for (int ct = 0; ct < 4; ++ct) {
            BF16x16 b;
            const unsigned short* bp = XtT + (size_t)(ct * 16 + mn) * NN + kb;
            b.h[0] = *(const us8*)bp; b.h[1] = *(const us8*)(bp + 16);
            ax[ct] = WMMA_BF16(a.v, b.v, ax[ct]);
        }
    }
#pragma unroll
    for (int r = 0; r < 8; ++r) {
        int i = ti * 16 + half * 8 + r;
        float rv = rs[i];
        float inv = (rv > 0.f) ? (1.f / rv) : 0.f;
#pragma unroll
        for (int ct = 0; ct < 4; ++ct)
            out[(size_t)i * DD + ct * 16 + mn] = ax[ct][r] * inv;
    }
}

// ------------------------------- host side ---------------------------------

extern "C" void kernel_launch(void* const* d_in, const int* in_sizes, int n_in,
                              void* d_out, int out_size, void* d_ws, size_t ws_size,
                              hipStream_t stream) {
    const float* Xs  = (const float*)d_in[0];   // [3,4096,64]
    const float* Ys  = (const float*)d_in[1];   // [3,4096,10]
    const float* Xt  = (const float*)d_in[2];   // [4096,64]
    const float* XB0 = (const float*)d_in[3];   // [4096,64]
    const float* YB0 = (const float*)d_in[4];   // [4096,10]
    float* out = (float*)d_out;                 // TXB [4096*64] ++ YB [4096*10]

    size_t off = 0;
    auto carve = [&](size_t bytes) -> void* {
        void* p = (char*)d_ws + off;
        off += (bytes + 255) & ~(size_t)255;
        return p;
    };
    float*          Cmat  = (float*)carve((size_t)NN * NN * 4);
    unsigned short* XsBf  = (unsigned short*)carve((size_t)KDOM * NN * DD * 2);
    unsigned short* YsBf  = (unsigned short*)carve((size_t)KDOM * NN * NCP * 2);
    unsigned short* XsT   = (unsigned short*)carve((size_t)KDOM * DD * NN * 2);
    unsigned short* YsT   = (unsigned short*)carve((size_t)KDOM * 16 * NN * 2);
    unsigned short* XBBf  = (unsigned short*)carve((size_t)NN * DD * 2);
    unsigned short* YBBf  = (unsigned short*)carve((size_t)NN * NCP * 2);
    unsigned short* XtBf  = (unsigned short*)carve((size_t)NN * DD * 2);
    unsigned short* XtT   = (unsigned short*)carve((size_t)DD * NN * 2);
    float*          XBcur = (float*)carve((size_t)NN * DD * 4);
    float*          YBcur = (float*)carve((size_t)NN * 16 * 4);
    float*          XBacc = (float*)carve((size_t)NN * DD * 4);
    float*          YBacc = (float*)carve((size_t)NN * 16 * 4);
    float*          fvec  = (float*)carve(NN * 4);
    float*          gvec  = (float*)carve(NN * 4);
    float*          rsv   = (float*)carve(NN * 4);
    float*          nXB   = (float*)carve(NN * 4);
    float*          nYB   = (float*)carve(NN * 4);
    float*          nXt   = (float*)carve(NN * 4);
    float*          nXs   = (float*)carve((size_t)KDOM * NN * 4);
    float*          nYs   = (float*)carve((size_t)KDOM * NN * 4);
    int*            scal  = (int*)carve(64);    // [0],[1]=max bits; [2],[3]=barrier

    const int DIST_BLOCKS = (NN / 16) * (NN / 16) / 4 / 8;   // TJ=4, 8 waves/block

    // ---- setup: bf16 conversions, norms, transposes (inputs are static) ----
    for (int k = 0; k < KDOM; ++k) {
        cvt_pad_norm_kernel<<<NN / 8, 256, 0, stream>>>(
            Xs + (size_t)k * NN * DD, DD, XsBf + (size_t)k * NN * DD, DD, nXs + (size_t)k * NN);
        cvt_pad_norm_kernel<<<NN / 8, 256, 0, stream>>>(
            Ys + (size_t)k * NN * NCL, NCL, YsBf + (size_t)k * NN * NCP, NCP, nYs + (size_t)k * NN);
        transpose_cvt_kernel<<<NN * DD / 256, 256, 0, stream>>>(
            Xs + (size_t)k * NN * DD, DD, XsT + (size_t)k * DD * NN, DD);
        transpose_cvt_kernel<<<NN * 16 / 256, 256, 0, stream>>>(
            Ys + (size_t)k * NN * NCL, NCL, YsT + (size_t)k * 16 * NN, 16);
    }
    cvt_pad_norm_kernel<<<NN / 8, 256, 0, stream>>>(Xt, DD, XtBf, DD, nXt);
    transpose_cvt_kernel<<<NN * DD / 256, 256, 0, stream>>>(Xt, DD, XtT, DD);
    (void)hipMemcpyAsync(XBcur, XB0, (size_t)NN * DD * 4, hipMemcpyDeviceToDevice, stream);
    pad_labels_f32_kernel<<<NN / 256, 256, 0, stream>>>(YB0, YBcur, NN);

    const float wn = (float)NN / (float)KDOM;   // w * n

    // ---- barycenter iterations ----
    for (int bi = 0; bi < NITER_BAR; ++bi) {
        cvt_pad_norm_kernel<<<NN / 8, 256, 0, stream>>>(XBcur, DD, XBBf, DD, nXB);
        cvt_pad_norm_kernel<<<NN / 8, 256, 0, stream>>>(YBcur, 16, YBBf, NCP, nYB);
        zero_kernel<<<NN * DD / 256, 256, 0, stream>>>(XBacc, NN * DD);
        zero_kernel<<<NN * 16 / 256, 256, 0, stream>>>(YBacc, NN * 16);
        for (int k = 0; k < KDOM; ++k) {
            reset_scal_kernel<<<1, 32, 0, stream>>>(scal);
            dist_wmma_kernel<DD, 4, false><<<DIST_BLOCKS, 256, 0, stream>>>(
                XBBf, nXB, XsBf + (size_t)k * NN * DD, nXs + (size_t)k * NN,
                Cmat, &scal[0], nullptr, nullptr);
            dist_wmma_kernel<NCP, 4, true><<<DIST_BLOCKS, 256, 0, stream>>>(
                YBBf, nYB, YsBf + (size_t)k * NN * NCP, nYs + (size_t)k * NN,
                Cmat, &scal[1], Cmat, &scal[0]);
            sinkhorn_kernel<<<SGRID, 256, 0, stream>>>(Cmat, fvec, gvec, &scal[1], &scal[2], NITER_SINK);
            proj_wmma_kernel<<<NN / 16 / 8, 256, 0, stream>>>(
                Cmat, fvec, gvec, &scal[1],
                XsT + (size_t)k * DD * NN, YsT + (size_t)k * 16 * NN, XBacc, YBacc);
        }
        scale_kernel<<<NN * DD / 256, 256, 0, stream>>>(XBacc, XBcur, wn, NN * DD);
        scale_kernel<<<NN * 16 / 256, 256, 0, stream>>>(YBacc, YBcur, wn, NN * 16);
    }

    // ---- final: transport barycenter -> target ----
    cvt_pad_norm_kernel<<<NN / 8, 256, 0, stream>>>(XBcur, DD, XBBf, DD, nXB);
    reset_scal_kernel<<<1, 32, 0, stream>>>(scal);
    dist_wmma_kernel<DD, 4, false><<<DIST_BLOCKS, 256, 0, stream>>>(
        XBBf, nXB, XtBf, nXt, Cmat, &scal[0], nullptr, nullptr);
    sinkhorn_kernel<<<SGRID, 256, 0, stream>>>(Cmat, fvec, gvec, &scal[0], &scal[2], NITER_SINK);
    rowsum_kernel<<<NN / 8, 256, 0, stream>>>(Cmat, fvec, gvec, &scal[0], rsv);
    trans_wmma_kernel<<<NN / 16 / 8, 256, 0, stream>>>(Cmat, fvec, gvec, &scal[0], XtT, rsv, out);
    copy_labels_out_kernel<<<NN / 256, 256, 0, stream>>>(YBcur, out + (size_t)NN * DD, NN);
}
